// GNN_44710609551751
// MI455X (gfx1250) — matrix operand order repaired and checked by reading, not compiled, not run on hardware
//
#include <hip/hip_runtime.h>
#include <hip/hip_bf16.h>

typedef __attribute__((ext_vector_type(16))) _Float16 v16h;
typedef __attribute__((ext_vector_type(8)))  float    v8f;
typedef __attribute__((ext_vector_type(4)))  unsigned int v4u;
typedef __attribute__((ext_vector_type(8)))  int      v8i;
typedef __attribute__((ext_vector_type(4)))  int      v4i;

#define H_DIM     128
#define S_DIM     256
#define BATCH_BD  1024                 // B*D = 16*64
#define M_ROWS    (BATCH_BD * S_DIM)   // 262144
#define N_LOOP    10

#if __has_builtin(__builtin_amdgcn_tensor_load_to_lds) && __has_builtin(__builtin_amdgcn_s_wait_tensorcnt)
#define USE_TDM 1
#else
#define USE_TDM 0
#endif

// ---------------------------------------------------------------------------
// Elementwise / prep kernels
// ---------------------------------------------------------------------------
__global__ void k_convert_f16(const float* __restrict__ x, _Float16* __restrict__ y, int n) {
    int i = blockIdx.x * blockDim.x + threadIdx.x;
    if (i < n) y[i] = (_Float16)x[i];
}

// in: [R, C] row-major f32  ->  out: [C, R] row-major f16   (for w_ih / w_hh transpose)
__global__ void k_transpose_f16(const float* __restrict__ x, _Float16* __restrict__ y, int R, int C) {
    int i = blockIdx.x * blockDim.x + threadIdx.x;
    if (i < R * C) {
        int r = i / C, c = i - r * C;
        y[c * R + r] = (_Float16)x[i];
    }
}

// mlp_w3 [128,1] -> padded f16 [128,64]; b3 [1] -> padded f32 [64]
__global__ void k_pad_w3(const float* __restrict__ w3, const float* __restrict__ b3,
                         _Float16* __restrict__ w3p, float* __restrict__ b3p) {
    int i = blockIdx.x * blockDim.x + threadIdx.x;
    if (i < 128 * 64) {
        int k = i >> 6, n = i & 63;
        w3p[i] = (_Float16)((n == 0) ? w3[k] : 0.0f);
    }
    if (i < 64) b3p[i] = (i == 0) ? b3[0] : 0.0f;
}

__global__ void k_gather_embed(const int* __restrict__ idx, const float* __restrict__ embed,
                               float* __restrict__ nodeF, _Float16* __restrict__ nodeH, int n) {
    int i = blockIdx.x * blockDim.x + threadIdx.x;
    if (i >= n) return;
    int m = i >> 7, h = i & 127;
    float v = embed[(long long)idx[m] * H_DIM + h];
    nodeF[i] = v;
    nodeH[i] = (_Float16)v;
}

__global__ void k_gru(const float* __restrict__ gi, const float* __restrict__ gh,
                      float* __restrict__ nodeF, _Float16* __restrict__ nodeH, int n) {
    int i = blockIdx.x * blockDim.x + threadIdx.x;
    if (i >= n) return;
    int m = i >> 7, h = i & 127;
    const float* gim = gi + (long long)m * 384;
    const float* ghm = gh + (long long)m * 384;
    float ir = gim[h],        hr = ghm[h];
    float iz = gim[128 + h],  hz = ghm[128 + h];
    float in_ = gim[256 + h], hn = ghm[256 + h];
    float r = 1.0f / (1.0f + expf(-(ir + hr)));
    float z = 1.0f / (1.0f + expf(-(iz + hz)));
    float nn = tanhf(in_ + r * hn);
    float ho = nodeF[i];
    float v = (1.0f - z) * nn + z * ho;
    nodeF[i] = v;
    nodeH[i] = (_Float16)v;
}

__global__ void k_extract_col0(const float* __restrict__ src, float* __restrict__ dst, int n) {
    int i = blockIdx.x * blockDim.x + threadIdx.x;
    if (i < n) dst[i] = src[(long long)i * 64];
}

// ---------------------------------------------------------------------------
// TDM helper: issue a 2D f16 tile DMA (tileW elems/row, tileH rows, row pitch
// ldElems) from global into LDS at byte offset ldsOff.  D# per ISA §8.3-8.6.
// ---------------------------------------------------------------------------
#if USE_TDM
__device__ __forceinline__ unsigned lds_offset(const void* p) {
    const __attribute__((address_space(3))) char* p3 =
        (const __attribute__((address_space(3))) char*)(const char*)p;
    return (unsigned)(unsigned long long)p3;
}

__device__ __forceinline__ void tdm_load_2d_f16(const _Float16* g, unsigned ldsOff,
                                                unsigned tileW, unsigned tileH,
                                                unsigned long long ldElems) {
    unsigned long long ga = (unsigned long long)(size_t)g;
    v4u g0;
    g0[0] = 1u;                                              // count=1, user descriptor
    g0[1] = ldsOff;                                          // lds_addr
    g0[2] = (unsigned)ga;                                    // global_addr[31:0]
    g0[3] = (unsigned)((ga >> 32) & 0x01FFFFFFu) | 0x80000000u; // addr[56:32] | type=2
    unsigned long long td0 = tileW, td1 = tileH, s0 = ldElems;
    v8i g1 = {0, 0, 0, 0, 0, 0, 0, 0};
    g1[0] = (int)(1u << 16);                                 // data_size=1 (2B), wg_mask=0
    g1[1] = (int)((unsigned)(td0 & 0xFFFFu) << 16);          // tensor_dim0[15:0]
    g1[2] = (int)((unsigned)((td0 >> 16) & 0xFFFFu) |
                  ((unsigned)(td1 & 0xFFFFu) << 16));        // tensor_dim0[31:16] | dim1[15:0]
    g1[3] = (int)((unsigned)((td1 >> 16) & 0xFFFFu) |
                  (tileW << 16));                            // tensor_dim1[31:16] | tile_dim0
    g1[4] = (int)(tileH & 0xFFFFu);                          // tile_dim1 (tile_dim2=0)
    g1[5] = (int)(unsigned)(s0 & 0xFFFFFFFFu);               // tensor_dim0_stride[31:0]
    g1[6] = (int)(unsigned)((s0 >> 32) & 0xFFFFu);           // stride[47:32] (dim1_stride=0)
    g1[7] = 0;
    v4i z4 = {0, 0, 0, 0};                                   // groups 2/3 unused (2D tensor)
    v8i z8 = {0, 0, 0, 0, 0, 0, 0, 0};
    __builtin_amdgcn_tensor_load_to_lds(g0, g1, z4, z4, z8, 0);  // clang-23: 6-arg form
}
#endif

// ---------------------------------------------------------------------------
// Tiled WMMA GEMM:  C[M,N] = act(A[M,K] * B[K,N] + bias)
// 128 threads = 4 waves (2x2); block tile 128Mx64N; K chunk 32.
// Each wave: 64Mx32N strip = 8 accumulators -> 8 x v_wmma_f32_16x16x32_f16
// per K chunk, with B fragments reused across 4 A fragments.
// LDS tiles are double-buffered and filled by the Tensor Data Mover (wave 0):
// chunk c+1's DMA flies while chunk c's WMMAs execute; s_wait_tensorcnt(2)
// blocks only on the older chunk (TDM completes in order per wave).
// ---------------------------------------------------------------------------
template<bool TANH>
__global__ __launch_bounds__(128)
void k_gemm_wmma(const _Float16* __restrict__ A, const _Float16* __restrict__ B,
                 const float* __restrict__ bias,
                 float* __restrict__ Cf, _Float16* __restrict__ Ch,
                 int M, int N, int K,
                 long long strideA, long long strideB, long long strideC) {
    __shared__ __align__(16) _Float16 lA[2][128 * 32];
    __shared__ __align__(16) _Float16 lB[2][32 * 64];

    const int tid   = threadIdx.x;
    const int wave  = tid >> 5;
    const int lane  = tid & 31;
    const int hlf   = lane >> 4;
    const int l16   = lane & 15;
    const int waveM = wave & 1;     // 2 waves along M
    const int waveN = wave >> 1;    // 2 waves along N

    const long long zA = (long long)blockIdx.z * strideA;
    const long long zB = (long long)blockIdx.z * strideB;
    const long long zC = (long long)blockIdx.z * strideC;
    const int rowBase = blockIdx.x * 128;
    const int colBase = blockIdx.y * 64;

    const _Float16* gA = A + zA + (long long)rowBase * K;   // + c*32 per chunk
    const _Float16* gB = B + zB + colBase;                  // + c*32*N per chunk

    const v8f vzero = {0.f, 0.f, 0.f, 0.f, 0.f, 0.f, 0.f, 0.f};
    v8f acc[4][2] = {{vzero, vzero}, {vzero, vzero}, {vzero, vzero}, {vzero, vzero}};

    const int nch = K >> 5;

#if USE_TDM
    const unsigned offA[2] = {lds_offset(lA[0]), lds_offset(lA[1])};
    const unsigned offB[2] = {lds_offset(lB[0]), lds_offset(lB[1])};
    if (wave == 0) {                                        // preload chunk 0 -> buffer 0
        tdm_load_2d_f16(gA, offA[0], 32u, 128u, (unsigned long long)K);
        tdm_load_2d_f16(gB, offB[0], 64u, 32u, (unsigned long long)N);
    }
#endif

    for (int c = 0; c < nch; ++c) {
        const int cur = c & 1;
        __syncthreads();            // readers of buf[cur^1] (iter c-1) are done
#if USE_TDM
        if (wave == 0) {
            if (c + 1 < nch) {      // prefetch next chunk into the other buffer
                tdm_load_2d_f16(gA + (c + 1) * 32, offA[cur ^ 1], 32u, 128u,
                                (unsigned long long)K);
                tdm_load_2d_f16(gB + (long long)(c + 1) * 32 * N, offB[cur ^ 1], 64u, 32u,
                                (unsigned long long)N);
                __builtin_amdgcn_s_wait_tensorcnt(2);   // older chunk (c) retired
            } else {
                __builtin_amdgcn_s_wait_tensorcnt(0);   // last chunk fully landed
            }
        }
#else
        {
            const int k0 = c * 32;
            #pragma unroll
            for (int i = tid; i < 512; i += 128) {          // A: 128x32 halves
                int r = i >> 2, s = i & 3;
                reinterpret_cast<uint4*>(lA[cur])[i] =
                    *reinterpret_cast<const uint4*>(gA + (long long)r * K + k0 + s * 8);
            }
            #pragma unroll
            for (int i = tid; i < 256; i += 128) {          // B: 32x64 halves
                int r = i >> 3, s = i & 7;
                reinterpret_cast<uint4*>(lB[cur])[i] =
                    *reinterpret_cast<const uint4*>(gB + (long long)(k0 + r) * N + s * 8);
            }
            if (k0 + 32 < K)
                __builtin_prefetch(gA + (long long)(tid >> 1) * K + k0 + 32, 0, 1);
        }
#endif
        __syncthreads();            // chunk c published to all waves

        const _Float16* pA = lA[cur];
        const _Float16* pB = lB[cur];

        // B fragments (32x16, K = 16*hlf + e, N = l16) -- 2 per wave, reused 4x
        v16h bf[2];
        #pragma unroll
        for (int ni = 0; ni < 2; ++ni) {
            const int col = waveN * 32 + ni * 16 + l16;
            #pragma unroll
            for (int e = 0; e < 16; ++e)
                bf[ni][e] = pB[(hlf * 16 + e) * 64 + col];
        }
        // A fragments (16x32, K = 16*hlf + e) -> pairs of ds_load_b128
        #pragma unroll
        for (int mi = 0; mi < 4; ++mi) {
            const int ar = waveM * 64 + mi * 16 + l16;
            v16h a;
            #pragma unroll
            for (int e = 0; e < 16; ++e) {
                int g = e >> 3, j = e & 7;
                a[e] = pA[ar * 32 + hlf * 16 + g * 8 + j];
            }
            #pragma unroll
            for (int ni = 0; ni < 2; ++ni)
                acc[mi][ni] = __builtin_amdgcn_wmma_f32_16x16x32_f16(
                    false, a, false, bf[ni], (short)0, acc[mi][ni], false, false);
        }
    }

    // Epilogue: C/D layout row = 8*hlf + v, col = l16
    #pragma unroll
    for (int mi = 0; mi < 4; ++mi) {
        #pragma unroll
        for (int ni = 0; ni < 2; ++ni) {
            int col = colBase + waveN * 32 + ni * 16 + l16;
            float bv = bias ? bias[col] : 0.0f;
            #pragma unroll
            for (int v = 0; v < 8; ++v) {
                int row = rowBase + waveM * 64 + mi * 16 + hlf * 8 + v;
                float val = acc[mi][ni][v] + bv;
                if (TANH) val = tanhf(val);
                long long off = zC + (long long)row * N + col;
                if (Cf) Cf[off] = val;
                if (Ch) Ch[off] = (_Float16)val;
            }
        }
    }
}

// ---------------------------------------------------------------------------
// Host orchestration
// ---------------------------------------------------------------------------
static inline dim3 gemm_grid(int M, int N, int Z) { return dim3(M / 128, N / 64, Z); }

extern "C" void kernel_launch(void* const* d_in, const int* in_sizes, int n_in,
                              void* d_out, int out_size, void* d_ws, size_t ws_size,
                              hipStream_t stream) {
    (void)in_sizes; (void)n_in; (void)out_size; (void)ws_size;
    const int*   input_var = (const int*)  d_in[0];
    const float* adjacency = (const float*)d_in[1];
    const float* embed     = (const float*)d_in[2];
    const float* gnn_w     = (const float*)d_in[3];
    const float* gnn_b     = (const float*)d_in[4];
    const float* w_ih      = (const float*)d_in[5];
    const float* w_hh      = (const float*)d_in[6];
    const float* b_ih      = (const float*)d_in[7];
    const float* b_hh      = (const float*)d_in[8];
    const float* mlp_w1    = (const float*)d_in[9];
    const float* mlp_b1    = (const float*)d_in[10];
    const float* mlp_w2    = (const float*)d_in[11];
    const float* mlp_b2    = (const float*)d_in[12];
    const float* mlp_w3    = (const float*)d_in[13];
    const float* mlp_b3    = (const float*)d_in[14];
    float* out = (float*)d_out;

    char* ws = (char*)d_ws;
    size_t off = 0;
    auto carve = [&](size_t bytes) -> char* {
        char* p = ws + off;
        off += (bytes + 255) & ~(size_t)255;
        return p;
    };
    const size_t MH = (size_t)M_ROWS * H_DIM;
    _Float16* nodeH = (_Float16*)carve(MH * 2);
    _Float16* tmpH  = (_Float16*)carve(MH * 2);               // reused as MLP t1
    _Float16* aggH  = (_Float16*)carve(MH * 2);               // reused as MLP t2
    _Float16* adjH  = (_Float16*)carve((size_t)BATCH_BD * S_DIM * S_DIM * 2);
    float*    nodeF = (float*)   carve(MH * 4);
    float*    giF   = (float*)   carve((size_t)M_ROWS * 384 * 4);
    float*    ghF   = (float*)   carve((size_t)M_ROWS * 384 * 4);
    float*    out64 = (float*)   carve((size_t)M_ROWS * 64 * 4);
    _Float16* gnnwH = (_Float16*)carve(128 * 128 * 2);
    _Float16* wihT  = (_Float16*)carve(128 * 384 * 2);
    _Float16* whhT  = (_Float16*)carve(128 * 384 * 2);
    _Float16* w1H   = (_Float16*)carve(128 * 128 * 2);
    _Float16* w2H   = (_Float16*)carve(128 * 128 * 2);
    _Float16* w3P   = (_Float16*)carve(128 * 64 * 2);
    float*    b3P   = (float*)   carve(64 * 4);

    const int TB = 256;
    auto blocks = [&](long long n) { return (unsigned)((n + TB - 1) / TB); };

    // Weight prep
    k_convert_f16<<<blocks(128 * 128), TB, 0, stream>>>(gnn_w, gnnwH, 128 * 128);
    k_transpose_f16<<<blocks(384 * 128), TB, 0, stream>>>(w_ih, wihT, 384, 128);
    k_transpose_f16<<<blocks(384 * 128), TB, 0, stream>>>(w_hh, whhT, 384, 128);
    k_convert_f16<<<blocks(128 * 128), TB, 0, stream>>>(mlp_w1, w1H, 128 * 128);
    k_convert_f16<<<blocks(128 * 128), TB, 0, stream>>>(mlp_w2, w2H, 128 * 128);
    k_pad_w3<<<blocks(128 * 64), TB, 0, stream>>>(mlp_w3, mlp_b3, w3P, b3P);
    k_convert_f16<<<blocks((long long)BATCH_BD * S_DIM * S_DIM), TB, 0, stream>>>(
        adjacency, adjH, BATCH_BD * S_DIM * S_DIM);

    // node = embed[input_var]
    k_gather_embed<<<blocks(MH), TB, 0, stream>>>(input_var, embed, nodeF, nodeH, (int)MH);

    // GNN + GRU loop
    for (int it = 0; it < N_LOOP; ++it) {
        k_gemm_wmma<false><<<gemm_grid(M_ROWS, 128, 1), 128, 0, stream>>>(
            nodeH, gnnwH, gnn_b, nullptr, tmpH, M_ROWS, 128, 128, 0, 0, 0);
        k_gemm_wmma<false><<<gemm_grid(S_DIM, 128, BATCH_BD), 128, 0, stream>>>(
            adjH, tmpH, nullptr, nullptr, aggH, S_DIM, 128, S_DIM,
            (long long)S_DIM * S_DIM, (long long)S_DIM * H_DIM, (long long)S_DIM * H_DIM);
        k_gemm_wmma<false><<<gemm_grid(M_ROWS, 384, 1), 128, 0, stream>>>(
            aggH, wihT, b_ih, giF, nullptr, M_ROWS, 384, 128, 0, 0, 0);
        k_gemm_wmma<false><<<gemm_grid(M_ROWS, 384, 1), 128, 0, stream>>>(
            nodeH, whhT, b_hh, ghF, nullptr, M_ROWS, 384, 128, 0, 0, 0);
        k_gru<<<blocks(MH), TB, 0, stream>>>(giF, ghF, nodeF, nodeH, (int)MH);
    }

    // MLP head
    k_gemm_wmma<true><<<gemm_grid(M_ROWS, 128, 1), 128, 0, stream>>>(
        nodeH, w1H, mlp_b1, nullptr, tmpH, M_ROWS, 128, 128, 0, 0, 0);
    k_gemm_wmma<true><<<gemm_grid(M_ROWS, 128, 1), 128, 0, stream>>>(
        tmpH, w2H, mlp_b2, nullptr, aggH, M_ROWS, 128, 128, 0, 0, 0);
    k_gemm_wmma<true><<<gemm_grid(M_ROWS, 64, 1), 128, 0, stream>>>(
        aggH, w3P, b3P, out64, nullptr, M_ROWS, 64, 128, 0, 0, 0);
    k_extract_col0<<<blocks(M_ROWS), TB, 0, stream>>>(out64, out, M_ROWS);
}
